// ANODE_46823733461182
// MI455X (gfx1250) — matrix operand order repaired and checked by reading, not compiled, not run on hardware
//
#include <hip/hip_runtime.h>

// ---------------------------------------------------------------------------
// ANODE Dopri5 integrator for MI455X (gfx1250, wave32, WMMA).
// One persistent kernel, fully on-chip:
//  - ALL weights (W1, W2, W3) live as per-wave WMMA B-fragments in the
//    gfx1250 extended VGPR file (256 VGPRs/wave of bf16 fragments, loaded
//    once from global fp32). Weights are time-invariant and each wave only
//    ever needs its own N-tiles, so every GEMM layer issues A-side LDS
//    loads only: 40 ds_load_b128 per wave per MLP eval, which balances the
//    64-bank LDS against the WMMA pipe.
//  - ODE state y and all six RK stage derivatives live in fp32 WMMA
//    C-fragments in registers; only the stage input Z round-trips through
//    LDS (C-layout -> A-layout transpose), H1/H2 round-trip as bf16.
//  - LDS use is just 20.5KB (Z + H1 + H2).
// Grid: 128 WGs x 256 threads (8 waves). Each WG owns 16 batch rows.
// ---------------------------------------------------------------------------

#define T_STEPS  128
#define BATCH    2048
#define DIM      128     // data(64) + aug(64)
#define WIDTH    256
#define BM       16      // batch rows per workgroup
#define NTHREADS 256

typedef __bf16 bf16;
typedef bf16  v16bf __attribute__((ext_vector_type(16)));
typedef bf16  v8bf  __attribute__((ext_vector_type(8)));
typedef float v8f   __attribute__((ext_vector_type(8)));
typedef float v4f   __attribute__((ext_vector_type(4)));

// LDS layout (bytes). Total 20480 bytes.
#define OFF_Z    0          // 16x128 bf16 = 4096
#define OFF_H1   4096       // 16x256 bf16 = 8192
#define OFF_H2   12288      // 16x256 bf16 = 8192
#define SMEM_BYTES 20480

__device__ __forceinline__ v8f wmma_bf16(v16bf a, v16bf b, v8f c) {
  // 8 args: (neg_a, A, neg_b, B, c_mod, C, reuse_a, reuse_b)
  return __builtin_amdgcn_wmma_f32_16x16x32_bf16(false, a, false, b, (short)0, c,
                                                 false, false);
}

// A-matrix 16x32 bf16 fragment from row-major [16 x ldk] tile in LDS.
// Lanes 0-15: M=lane, K = 0..15; lanes 16-31: K = 16..31.
__device__ __forceinline__ v16bf load_a_frag(const bf16* tile, int ldk, int lane) {
  const int half = (lane >> 4) & 1;
  const int m    = lane & 15;
  const bf16* row = tile + m * ldk;
  v8bf lo = *reinterpret_cast<const v8bf*>(row + half * 16);
  v8bf hi = *reinterpret_cast<const v8bf*>(row + half * 16 + 8);
  return __builtin_shufflevector(lo, hi, 0, 1, 2, 3, 4, 5, 6, 7,
                                         8, 9, 10, 11, 12, 13, 14, 15);
}

// B-matrix 32x16 bf16 fragment built once from GLOBAL fp32 weights
// (row-major [N x ldk]); lives in VGPRs for the whole kernel.
// Lanes 0-15 hold K=0..15 of column n=lane&15; lanes 16-31 hold K=16..31.
__device__ __forceinline__ v16bf load_b_frag_global(const float* W, int ldk,
                                                    int nrow, int kbase,
                                                    int lane) {
  const int half = (lane >> 4) & 1;
  const int nn   = lane & 15;
  const float* p = W + (size_t)(nrow + nn) * ldk + kbase + half * 16;
  v4f a0 = *reinterpret_cast<const v4f*>(p);
  v4f a1 = *reinterpret_cast<const v4f*>(p + 4);
  v4f a2 = *reinterpret_cast<const v4f*>(p + 8);
  v4f a3 = *reinterpret_cast<const v4f*>(p + 12);
  v16bf f;
  #pragma unroll
  for (int i = 0; i < 4; ++i) {
    f[i]      = (bf16)a0[i];
    f[4 + i]  = (bf16)a1[i];
    f[8 + i]  = (bf16)a2[i];
    f[12 + i] = (bf16)a3[i];
  }
  return f;
}

// C/D 16x16 f32 fragment -> bf16 activation tile in LDS, fused bias + relu.
// VGPR r: lanes 0-15 -> M=r, lanes 16-31 -> M=8+r; N = ntile*16 + (lane&15).
__device__ __forceinline__ void store_act(bf16* H, int ld, int ntile, int lane,
                                          v8f c, float b) {
  const int half = (lane >> 4) & 1;
  const int n    = ntile * 16 + (lane & 15);
  #pragma unroll
  for (int r = 0; r < 8; ++r) {
    float val = fmaxf(c[r] + b, 0.0f);
    H[(half * 8 + r) * ld + n] = (bf16)val;
  }
}

// One MLP evaluation. Z (bf16 [16x128]) must already be written by all waves;
// returns this wave's k C-fragment (rows half*8+r, column n = 16*w+lane15),
// bias b3 included. No trailing barrier needed: the k fragment is private.
__device__ __forceinline__ v8f feval_frag(const v16bf (&w1f)[2][4],
                                          const v16bf (&w2f)[2][8],
                                          const v16bf (&w3f)[8],
                                          float b1a, float b1b,
                                          float b2a, float b2b, float b3v,
                                          const bf16* Z, bf16* H1, bf16* H2,
                                          int tid) {
  const int lane = tid & 31;
  const int w    = tid >> 5;
  __syncthreads();  // Z visible; also fences H1/H2 reuse from previous stage

  // Layer 1: H1[16,256] = relu(Z[16,128] @ W1^T + b1). 16 N-tiles / 8 waves.
  {
    v8f c0 = {0.f, 0.f, 0.f, 0.f, 0.f, 0.f, 0.f, 0.f};
    v8f c1 = {0.f, 0.f, 0.f, 0.f, 0.f, 0.f, 0.f, 0.f};
    #pragma unroll
    for (int kt = 0; kt < DIM / 32; ++kt) {
      v16bf a = load_a_frag(Z + kt * 32, DIM, lane);
      c0 = wmma_bf16(a, w1f[0][kt], c0);
      c1 = wmma_bf16(a, w1f[1][kt], c1);
    }
    store_act(H1, WIDTH, 2 * w, lane, c0, b1a);
    store_act(H1, WIDTH, 2 * w + 1, lane, c1, b1b);
  }
  __syncthreads();

  // Layer 2: H2[16,256] = relu(H1 @ W2^T + b2). 16 N-tiles / 8 waves.
  {
    v8f c0 = {0.f, 0.f, 0.f, 0.f, 0.f, 0.f, 0.f, 0.f};
    v8f c1 = {0.f, 0.f, 0.f, 0.f, 0.f, 0.f, 0.f, 0.f};
    #pragma unroll
    for (int kt = 0; kt < WIDTH / 32; ++kt) {
      v16bf a = load_a_frag(H1 + kt * 32, WIDTH, lane);
      c0 = wmma_bf16(a, w2f[0][kt], c0);
      c1 = wmma_bf16(a, w2f[1][kt], c1);
    }
    store_act(H2, WIDTH, 2 * w, lane, c0, b2a);
    store_act(H2, WIDTH, 2 * w + 1, lane, c1, b2b);
  }
  __syncthreads();

  // Layer 3: k[16,128] = H2 @ W3^T + b3, kept as C fragment. 8 tiles/8 waves.
  v8f c = {0.f, 0.f, 0.f, 0.f, 0.f, 0.f, 0.f, 0.f};
  {
    #pragma unroll
    for (int kt = 0; kt < WIDTH / 32; ++kt) {
      v16bf a = load_a_frag(H2 + kt * 32, WIDTH, lane);
      c = wmma_bf16(a, w3f[kt], c);
    }
    #pragma unroll
    for (int r = 0; r < 8; ++r) c[r] = c[r] + b3v;
  }
  return c;
}

// Write this wave's stage-input fragment into Z (bf16, row-major [16x128]).
__device__ __forceinline__ void write_z(bf16* Z, v8f zf, int half, int n) {
  #pragma unroll
  for (int r = 0; r < 8; ++r) Z[(half * 8 + r) * DIM + n] = (bf16)zf[r];
}

#define FEVAL() feval_frag(w1f, w2f, w3f, b1a, b1b, b2a, b2b, b3v, Z, H1, H2, tid)

__global__ __launch_bounds__(NTHREADS, 1)
void anode_dopri5_kernel(const float* __restrict__ ts,
                         const float* __restrict__ y0,
                         const float* __restrict__ W1, const float* __restrict__ b1,
                         const float* __restrict__ W2, const float* __restrict__ b2,
                         const float* __restrict__ W3, const float* __restrict__ b3,
                         float* __restrict__ out, int* __restrict__ nsteps) {
  extern __shared__ char smem[];
  bf16* Z  = (bf16*)(smem + OFF_Z);
  bf16* H1 = (bf16*)(smem + OFF_H1);
  bf16* H2 = (bf16*)(smem + OFF_H2);

  const int tid = threadIdx.x;
  const int wg  = blockIdx.x;

  if (wg == 0 && tid == 0) *nsteps = (T_STEPS - 1) * 4;  // 508

  const int lane = tid & 31;
  const int w    = tid >> 5;
  const int half = (lane >> 4) & 1;
  const int l15  = lane & 15;
  const int n    = w * 16 + l15;  // dim column owned by this wave

  // Per-wave time-invariant B-fragments of all three weight matrices,
  // loaded once from global fp32 into 256 VGPRs of bf16 fragments
  // (gfx1250 extended VGPR file via s_set_vgpr_msb).
  v16bf w1f[2][4];
  #pragma unroll
  for (int t = 0; t < 2; ++t)
    #pragma unroll
    for (int kt = 0; kt < DIM / 32; ++kt)
      w1f[t][kt] = load_b_frag_global(W1, DIM, (2 * w + t) * 16, kt * 32, lane);
  v16bf w2f[2][8];
  #pragma unroll
  for (int t = 0; t < 2; ++t)
    #pragma unroll
    for (int kt = 0; kt < WIDTH / 32; ++kt)
      w2f[t][kt] = load_b_frag_global(W2, WIDTH, (2 * w + t) * 16, kt * 32, lane);
  v16bf w3f[8];
  #pragma unroll
  for (int kt = 0; kt < WIDTH / 32; ++kt)
    w3f[kt] = load_b_frag_global(W3, WIDTH, w * 16, kt * 32, lane);

  // Per-lane bias registers (each lane only needs its own columns).
  const float b1a = b1[(2 * w) * 16 + l15];
  const float b1b = b1[(2 * w + 1) * 16 + l15];
  const float b2a = b2[(2 * w) * 16 + l15];
  const float b2b = b2[(2 * w + 1) * 16 + l15];
  const float b3v = b3[n];

  // Load y0 into the state C-fragment (rows half*8+r, column n).
  v8f yf;
  #pragma unroll
  for (int r = 0; r < 8; ++r)
    yf[r] = y0[wg * (BM * DIM) + (half * 8 + r) * DIM + n];

  // ys[0] = y0[:, :64]
  if (n < 64) {
    #pragma unroll
    for (int r = 0; r < 8; ++r)
      out[(size_t)(wg * BM + half * 8 + r) * 64 + n] = yf[r];
  }

  v8f k1, k2, k3, k4, k5, k6, z;

  for (int iv = 0; iv < T_STEPS - 1; ++iv) {
    const float dt = (ts[iv + 1] - ts[iv]) * 0.25f;  // / SUBSTEPS
    for (int s = 0; s < 4; ++s) {
      // stage 1
      write_z(Z, yf, half, n);
      k1 = FEVAL();
      // stage 2
      z = yf + dt * (0.2f * k1);
      write_z(Z, z, half, n);
      k2 = FEVAL();
      // stage 3
      z = yf + dt * ((3.f / 40.f) * k1 + (9.f / 40.f) * k2);
      write_z(Z, z, half, n);
      k3 = FEVAL();
      // stage 4
      z = yf + dt * ((44.f / 45.f) * k1 - (56.f / 15.f) * k2
                   + (32.f / 9.f) * k3);
      write_z(Z, z, half, n);
      k4 = FEVAL();
      // stage 5
      z = yf + dt * ((19372.f / 6561.f) * k1 - (25360.f / 2187.f) * k2
                   + (64448.f / 6561.f) * k3 - (212.f / 729.f) * k4);
      write_z(Z, z, half, n);
      k5 = FEVAL();
      // stage 6
      z = yf + dt * ((9017.f / 3168.f) * k1 - (355.f / 33.f) * k2
                   + (46732.f / 5247.f) * k3 + (49.f / 176.f) * k4
                   - (5103.f / 18656.f) * k5);
      write_z(Z, z, half, n);
      k6 = FEVAL();
      // 5th-order update
      yf = yf + dt * ((35.f / 384.f) * k1 + (500.f / 1113.f) * k3
                    + (125.f / 192.f) * k4 - (2187.f / 6784.f) * k5
                    + (11.f / 84.f) * k6);
    }
    // ys[iv+1][:, :64]
    if (n < 64) {
      #pragma unroll
      for (int r = 0; r < 8; ++r)
        out[((size_t)(iv + 1) * BATCH + wg * BM + half * 8 + r) * 64 + n] = yf[r];
    }
  }
}

extern "C" void kernel_launch(void* const* d_in, const int* in_sizes, int n_in,
                              void* d_out, int out_size, void* d_ws, size_t ws_size,
                              hipStream_t stream) {
  (void)in_sizes; (void)n_in; (void)d_ws; (void)ws_size;
  const float* ts = (const float*)d_in[0];
  const float* y0 = (const float*)d_in[1];
  const float* W1 = (const float*)d_in[2];
  const float* b1 = (const float*)d_in[3];
  const float* W2 = (const float*)d_in[4];
  const float* b2 = (const float*)d_in[5];
  const float* W3 = (const float*)d_in[6];
  const float* b3 = (const float*)d_in[7];
  float* out = (float*)d_out;
  int* nsteps = (int*)d_out + ((size_t)out_size - 1);  // int32 num_steps slot

  anode_dopri5_kernel<<<dim3(BATCH / BM), dim3(NTHREADS), SMEM_BYTES, stream>>>(
      ts, y0, W1, b1, W2, b2, W3, b3, out, nsteps);
}
